// GLMAttention_85796266705209
// MI455X (gfx1250) — matrix operand order repaired and checked by reading, not compiled, not run on hardware
//
#include <hip/hip_runtime.h>
#include <hip/hip_bf16.h>

// ---------------- CDNA5 WMMA plumbing ----------------
typedef __attribute__((ext_vector_type(16))) __bf16 v16bf;
typedef __attribute__((ext_vector_type(8)))  float  v8f;
typedef __attribute__((ext_vector_type(4)))  int    v4i;

static __device__ inline v8f wmma_bf16(v16bf a, v16bf b, v8f c) {
  // D = A(16x32 bf16) * B(32x16 bf16) + C(16x16 f32)
  return __builtin_amdgcn_wmma_f32_16x16x32_bf16(
      /*neg_a=*/false, a, /*neg_b=*/false, b,
      /*c_mod=*/(short)0, c, /*reuse_a=*/false, /*reuse_b=*/false);
}

// ---------------- gfx1250 async global->LDS (ASYNCcnt-tracked DMA) ----------------
// Probe (round 2 diagnostic) revealed the builtin's prototype: arg0 is
// 'int __vector(4) __device__ *' i.e. v4i in addrspace(1); arg1 is the LDS
// destination (addrspace(3)); then imm offset, imm cpol.
#if __has_builtin(__builtin_amdgcn_global_load_async_to_lds_b128)
#define HAVE_ASYNC_LDS 1
typedef __attribute__((address_space(1))) v4i gas_v4i;
typedef __attribute__((address_space(3))) v4i las_v4i;
#else
#define HAVE_ASYNC_LDS 0
#endif

static __device__ inline void stage_b128(const __bf16* g, __bf16* l) {
#if HAVE_ASYNC_LDS
  __builtin_amdgcn_global_load_async_to_lds_b128(
      (gas_v4i*)g, (las_v4i*)l, /*offset=*/0, /*cpol=*/0);
#else
  *(uint4*)l = *(const uint4*)g;
#endif
}

static __device__ inline void wait_async_all() {
#if HAVE_ASYNC_LDS
#if __has_builtin(__builtin_amdgcn_s_wait_asynccnt)
  __builtin_amdgcn_s_wait_asynccnt(0);
#else
  asm volatile("s_wait_asynccnt 0x0" ::: "memory");
#endif
#endif
}

// Problem constants (from reference)
#define TSEQ    2048
#define HIDDEN  4096
#define NQH     32
#define NKVH    2
#define HDIM    128
#define QSIZE   4096          // NQH*HDIM
#define KVSIZE  256           // NKVH*HDIM
#define QKVN    4608          // QSIZE + 2*KVSIZE
#define SCALE_Q 0.08838834764831845f   // 128^-0.5

// ---------------- Kernel 1: fp32 -> bf16 cast ----------------
__global__ __launch_bounds__(256) void cast_f32_bf16(
    const float* __restrict__ src, __bf16* __restrict__ dst, int n) {
  int i = (blockIdx.x * 256 + threadIdx.x) * 8;
  if (i >= n) return;
#pragma unroll
  for (int j = 0; j < 8; ++j) dst[i + j] = (__bf16)src[i + j];
}

// ---------------- Kernel 2/5: bf16 WMMA GEMM  C = A * B^T (+bias) ----------------
// A: [M][K] bf16 row-major, B: [N][K] bf16 row-major (weights), C: [M][N] f32.
// Block tile 128x128, 256 threads = 8 waves; K step 32.
// Double-buffered LDS; next K-tile staged via async-to-LDS DMA while WMMAs
// consume the current one (ASYNCcnt overlap), one asynccnt wait per K-step.
#define LDSS 40   // bf16 stride per LDS tile row (32 data + 8 pad); 80B keeps 16B alignment

__global__ __launch_bounds__(256) void gemm_bf16_kernel(
    const __bf16* __restrict__ A, const __bf16* __restrict__ B,
    const float* __restrict__ bias, float* __restrict__ C,
    int M, int N, int K) {
  __shared__ __align__(16) __bf16 lA[2][128 * LDSS];
  __shared__ __align__(16) __bf16 lB[2][128 * LDSS];

  const int tid  = threadIdx.x;
  const int wave = tid >> 5;
  const int lane = tid & 31;
  const int m0 = blockIdx.y * 128;
  const int n0 = blockIdx.x * 128;

  v8f acc[8] = {};

  const int row = tid >> 1;          // 0..127 (tile row loaded by this thread)
  const int seg = (tid & 1) * 16;    // 0 or 16 (16 bf16 = 32B per thread)
  const __bf16* ag = A + (size_t)(m0 + row) * K + seg;
  const __bf16* bg = B + (size_t)(n0 + row) * K + seg;
  const int ldst = row * LDSS + seg;

  // fragment addressing (contiguous k order: lane half h holds k = 16h..16h+15)
  const int arow = wave * 16 + (lane & 15);
  const int akb  = (lane & 16) ? 16 : 0;     // A: lanes16-31 hold K+16
  const int bn   = lane & 15;
  const int bk   = (lane & 16) ? 16 : 0;     // B: lanes16-31 hold K+16

  // prologue: stage K-tile 0 into buffer 0
  stage_b128(ag, &lA[0][ldst]);
  stage_b128(bg, &lB[0][ldst]);
  wait_async_all();
  __syncthreads();

  int buf = 0;
  for (int k0 = 0; k0 < K; k0 += 32) {
    // kick off DMA of the next K-tile into the other buffer
    if (k0 + 32 < K) {
      stage_b128(ag + k0 + 32, &lA[buf ^ 1][ldst]);
      stage_b128(bg + k0 + 32, &lB[buf ^ 1][ldst]);
    }

    // A fragment: lanes 0-15 row M=lane, K in 0..15; lanes 16-31 +16
    v16bf af;
#pragma unroll
    for (int e = 0; e < 16; ++e)
      af[e] = lA[buf][arow * LDSS + akb + e];

    // preload all 8 B fragments, then run the WMMAs as one clause
    v16bf bf[8];
#pragma unroll
    for (int ct = 0; ct < 8; ++ct)
#pragma unroll
      for (int e = 0; e < 16; ++e)
        bf[ct][e] = lB[buf][(ct * 16 + bn) * LDSS + bk + e];

#pragma unroll
    for (int ct = 0; ct < 8; ++ct)
      acc[ct] = wmma_bf16(af, bf[ct], acc[ct]);

    wait_async_all();     // next tile landed (this wave)
    __syncthreads();      // ... and all waves; also fences buf reuse
    buf ^= 1;
  }

  // epilogue: C/D layout -> VGPR v holds M=v (lanes<16) or M=v+8 (lanes>=16)
  const int crow = m0 + wave * 16 + ((lane & 16) ? 8 : 0);
#pragma unroll
  for (int ct = 0; ct < 8; ++ct) {
    const int col = n0 + ct * 16 + bn;
    const float bv = bias ? bias[col] : 0.0f;
#pragma unroll
    for (int v = 0; v < 8; ++v)
      C[(size_t)(crow + v) * N + col] = acc[ct][v] + bv;
  }
}

// ---------------- Kernel 3: RoPE + scale + cast to attention layouts ----------------
// qkv f32 [T][4608] -> q bf16 [T][32][128] (pre-scaled), k bf16 [T][2][128],
// vT bf16 [2][128][T] (transposed so P*V B-fragments are contiguous loads).
__global__ __launch_bounds__(128) void rope_cast_kernel(
    const float* __restrict__ qkv, const int* __restrict__ positions,
    __bf16* __restrict__ qb, __bf16* __restrict__ kb, __bf16* __restrict__ vT) {
  const int t = blockIdx.y;
  const int h = blockIdx.x;      // 0..35: 0-31 Q heads, 32-33 K heads, 34-35 V heads
  const int d = threadIdx.x;     // 0..127
  const float x = qkv[(size_t)t * QKVN + h * HDIM + d];
  float y = x;
  if (h < 34 && d < 64) {        // rotate first ROT_DIM=64 dims of Q and K
    const float pos = (float)positions[t];
    const int i = d >> 1;
    const float inv = __powf(10000.0f, -(float)i * (1.0f / 32.0f));
    const float f = pos * inv;
    const float c = __cosf(f), s = __sinf(f);
    const float other = __shfl_xor(x, 1);   // pair partner (lanes d, d^1 in same wave32)
    y = (d & 1) ? (x * c + other * s) : (x * c - other * s);
  }
  if (h < 32) {
    qb[((size_t)t * NQH + h) * HDIM + d] = (__bf16)(y * SCALE_Q);
  } else if (h < 34) {
    kb[((size_t)t * NKVH + (h - 32)) * HDIM + d] = (__bf16)y;
  } else {
    vT[((size_t)(h - 34) * HDIM + d) * TSEQ + t] = (__bf16)y;
  }
}

// ---------------- Kernel 4: causal flash attention (WMMA) ----------------
// One wave per (head, 16-row Q tile). S-blocks of 32 columns; 16 WMMAs per block.
__global__ __launch_bounds__(128) void flash_attn_kernel(
    const __bf16* __restrict__ qb, const __bf16* __restrict__ kb,
    const __bf16* __restrict__ vT, __bf16* __restrict__ attn) {
  __shared__ __align__(16) __bf16 Plds[4][16][LDSS];  // per-wave P staging (C->A layout)

  const int wave = threadIdx.x >> 5;
  const int lane = threadIdx.x & 31;
  const int h     = blockIdx.x;                 // 0..31
  const int qtile = blockIdx.y * 4 + wave;      // 0..127
  const int kvh   = h >> 4;                     // rep = 16
  const int qbase = qtile * 16;

  // Q A-fragments (16 rows x 128 d), loaded once: 4 fragments over d
  v16bf qf[4];
  {
    const int qrow = qbase + (lane & 15);
    const int akb = (lane & 16) ? 16 : 0;
#pragma unroll
    for (int kblk = 0; kblk < 4; ++kblk)
#pragma unroll
      for (int e = 0; e < 16; ++e) {
        const int dd = kblk * 32 + akb + e;
        qf[kblk][e] = qb[((size_t)qrow * NQH + h) * HDIM + dd];
      }
  }

  v8f o[8] = {};                 // O accumulator: 16 rows x 128 d
  float mrow[8], lrow[8];
#pragma unroll
  for (int v = 0; v < 8; ++v) { mrow[v] = -1e30f; lrow[v] = 0.0f; }

  const int rbase = qbase + ((lane & 16) ? 8 : 0);   // row of acc element v is rbase+v
  const int send = qbase + 16;
  for (int s0 = 0; s0 < send; s0 += 32) {
    // ---- S = Q * K^T : preload all 8 K B-fragments, then 8 WMMAs ----
    v16bf kf[2][4];
#pragma unroll
    for (int nt = 0; nt < 2; ++nt) {
      const int scol = s0 + nt * 16 + (lane & 15);
      const int bk = (lane & 16) ? 16 : 0;
#pragma unroll
      for (int kblk = 0; kblk < 4; ++kblk)
#pragma unroll
        for (int e = 0; e < 16; ++e)
          kf[nt][kblk][e] = kb[((size_t)scol * NKVH + kvh) * HDIM + kblk * 32 + bk + e];
    }
    v8f Sv[2] = {};
#pragma unroll
    for (int nt = 0; nt < 2; ++nt)
#pragma unroll
      for (int kblk = 0; kblk < 4; ++kblk)
        Sv[nt] = wmma_bf16(qf[kblk], kf[nt][kblk], Sv[nt]);

    // ---- causal mask + online softmax (row reductions within 16-lane half) ----
    float fac[8];
#pragma unroll
    for (int v = 0; v < 8; ++v) {
      const int r = rbase + v;
#pragma unroll
      for (int nt = 0; nt < 2; ++nt) {
        const int c = s0 + nt * 16 + (lane & 15);
        if (c > r) Sv[nt][v] = -1e30f;
      }
      float m = fmaxf(Sv[0][v], Sv[1][v]);
#pragma unroll
      for (int off = 8; off >= 1; off >>= 1) m = fmaxf(m, __shfl_xor(m, off));
      const float mnew = fmaxf(mrow[v], m);
      fac[v] = __expf(mrow[v] - mnew);
      mrow[v] = mnew;
      float rs = 0.0f;
#pragma unroll
      for (int nt = 0; nt < 2; ++nt) {
        const float p = __expf(Sv[nt][v] - mnew);
        rs += p;
        Plds[wave][((lane & 16) ? 8 : 0) + v][nt * 16 + (lane & 15)] = (__bf16)p;
      }
#pragma unroll
      for (int off = 8; off >= 1; off >>= 1) rs += __shfl_xor(rs, off);
      lrow[v] = lrow[v] * fac[v] + rs;
    }
#pragma unroll
    for (int dt = 0; dt < 8; ++dt)
#pragma unroll
      for (int v = 0; v < 8; ++v) o[dt][v] = o[dt][v] * fac[v];

    // ---- O += P * V : P from LDS (A layout), V fragments preloaded ----
    v16bf pf;
    {
      const int prow = lane & 15;
      const int akb = (lane & 16) ? 16 : 0;
#pragma unroll
      for (int e = 0; e < 16; ++e)
        pf[e] = Plds[wave][prow][akb + e];
    }
    v16bf vf[8];
#pragma unroll
    for (int dt = 0; dt < 8; ++dt) {
      const int dcol = dt * 16 + (lane & 15);
      const int sb = (lane & 16) ? 16 : 0;
#pragma unroll
      for (int e = 0; e < 16; ++e)
        vf[dt][e] = vT[((size_t)kvh * HDIM + dcol) * TSEQ + s0 + sb + e];
    }
#pragma unroll
    for (int dt = 0; dt < 8; ++dt)
      o[dt] = wmma_bf16(pf, vf[dt], o[dt]);
  }

  // ---- normalize and store attn (bf16, feeds dense GEMM) ----
#pragma unroll
  for (int dt = 0; dt < 8; ++dt) {
    const int dcol = h * HDIM + dt * 16 + (lane & 15);
#pragma unroll
    for (int v = 0; v < 8; ++v)
      attn[(size_t)(rbase + v) * QSIZE + dcol] = (__bf16)(o[dt][v] / lrow[v]);
  }
}

// ---------------- host-side orchestration ----------------
extern "C" void kernel_launch(void* const* d_in, const int* in_sizes, int n_in,
                              void* d_out, int out_size, void* d_ws, size_t ws_size,
                              hipStream_t stream) {
  const float* hidden   = (const float*)d_in[0];
  const int*   positions= (const int*)d_in[1];
  const float* w_qkv    = (const float*)d_in[2];
  const float* b_qkv    = (const float*)d_in[3];
  const float* w_dense  = (const float*)d_in[4];
  float* out = (float*)d_out;
  char* ws = (char*)d_ws;

  // workspace layout (bytes)
  const size_t OFF_HS   = 0;                                     // T*H bf16
  const size_t OFF_WQKV = OFF_HS   + (size_t)TSEQ * HIDDEN * 2;  // QKVN*H bf16
  const size_t OFF_WD   = OFF_WQKV + (size_t)QKVN * HIDDEN * 2;  // H*H bf16
  const size_t OFF_QKV  = OFF_WD   + (size_t)HIDDEN * HIDDEN * 2;// T*QKVN f32
  const size_t OFF_Q    = OFF_QKV  + (size_t)TSEQ * QKVN * 4;    // T*QSIZE bf16
  const size_t OFF_K    = OFF_Q    + (size_t)TSEQ * QSIZE * 2;   // T*KVSIZE bf16
  const size_t OFF_VT   = OFF_K    + (size_t)TSEQ * KVSIZE * 2;  // KVSIZE*T bf16
  const size_t OFF_ATTN = OFF_VT   + (size_t)TSEQ * KVSIZE * 2;  // T*QSIZE bf16

  __bf16* hs_bf   = (__bf16*)(ws + OFF_HS);
  __bf16* wqkv_bf = (__bf16*)(ws + OFF_WQKV);
  __bf16* wd_bf   = (__bf16*)(ws + OFF_WD);
  float*  qkv_f32 = (float*) (ws + OFF_QKV);
  __bf16* q_bf    = (__bf16*)(ws + OFF_Q);
  __bf16* k_bf    = (__bf16*)(ws + OFF_K);
  __bf16* vT_bf   = (__bf16*)(ws + OFF_VT);
  __bf16* attn_bf = (__bf16*)(ws + OFF_ATTN);

  auto blocks8 = [](long long n) { return (unsigned)((n / 8 + 255) / 256); };

  // 1) fp32 -> bf16 casts
  cast_f32_bf16<<<blocks8((long long)TSEQ * HIDDEN), 256, 0, stream>>>(
      hidden, hs_bf, TSEQ * HIDDEN);
  cast_f32_bf16<<<blocks8((long long)QKVN * HIDDEN), 256, 0, stream>>>(
      w_qkv, wqkv_bf, QKVN * HIDDEN);
  cast_f32_bf16<<<blocks8((long long)HIDDEN * HIDDEN), 256, 0, stream>>>(
      w_dense, wd_bf, HIDDEN * HIDDEN);

  // 2) QKV GEMM + bias (f32 out)
  gemm_bf16_kernel<<<dim3(QKVN / 128, TSEQ / 128), 256, 0, stream>>>(
      hs_bf, wqkv_bf, b_qkv, qkv_f32, TSEQ, QKVN, HIDDEN);

  // 3) RoPE + scale + cast to q/k/vT bf16
  rope_cast_kernel<<<dim3(36, TSEQ), 128, 0, stream>>>(
      qkv_f32, positions, q_bf, k_bf, vT_bf);

  // 4) causal flash attention -> attn bf16
  flash_attn_kernel<<<dim3(NQH, TSEQ / 16 / 4), 128, 0, stream>>>(
      q_bf, k_bf, vT_bf, attn_bf);

  // 5) dense projection -> f32 output
  gemm_bf16_kernel<<<dim3(QSIZE / 128, TSEQ / 128), 256, 0, stream>>>(
      attn_bf, wd_bf, nullptr, out, TSEQ, QSIZE, HIDDEN);
}
